// RotaryAttention_89618787598958
// MI455X (gfx1250) — compile-verified, matrix operand order checked by reading, not run on hardware
//
#include <hip/hip_runtime.h>
#include <hip/hip_bf16.h>

typedef __bf16 bf16;
typedef __attribute__((ext_vector_type(16))) __bf16 v16bf;
typedef __attribute__((ext_vector_type(8)))  float  v8f;

#define T_LEN 4096
#define D_MODEL 1024
#define N_HEADS 16
#define HEAD_DIM 64
#define NQKV 3072

// ---------------------------------------------------------------------------
// Fragment loaders matching the CDNA5 WMMA VGPR layouts (CDNA5 ISA)
// A operand (16x32 bf16, row-major src, stride ld):
//   lane l: row = l&15, elements 0..7 = K[ks..ks+7], 8..15 = K[16+ks..16+ks+7],
//   ks = (l>>4)*8.  Same layout serves the B operand when the source is B^T
//   stored row-major [N][K].
// ---------------------------------------------------------------------------
__device__ __forceinline__ v16bf ldfrag(const bf16* p, int ld) {
  int lane = threadIdx.x & 31;
  const bf16* r = p + (size_t)(lane & 15) * ld + ((lane >> 4) << 3);
  union { v16bf v; uint2 q[4]; } u;
  u.q[0] = *(const uint2*)(r);
  u.q[1] = *(const uint2*)(r + 4);
  u.q[2] = *(const uint2*)(r + 16);
  u.q[3] = *(const uint2*)(r + 20);
  return u.v;
}

__device__ __forceinline__ v8f wmma_bf16(v16bf a, v16bf b, v8f c) {
  return __builtin_amdgcn_wmma_f32_16x16x32_bf16(
      /*neg_a=*/false, a, /*neg_b=*/false, b,
      /*c_mod=*/(short)0, c, /*reuse_a=*/false, /*reuse_b=*/false);
}

// Async global->LDS B128 (ISA §10: GLOBAL_LOAD_ASYNC_TO_LDS_B128, ASYNCcnt).
// lds_off: wave-relative LDS byte offset (low 32 bits of generic shared ptr).
__device__ __forceinline__ void async_b128(unsigned lds_off, const bf16* gptr) {
  asm volatile("global_load_async_to_lds_b128 %0, %1, off"
               :: "v"(lds_off), "v"(gptr) : "memory");
}
__device__ __forceinline__ unsigned lds_off32(const void* p) {
  return (unsigned)(unsigned long long)p;   // generic LDS addr[31:0] = LDS offset
}

// ---------------------------------------------------------------------------
// fp32 -> bf16 conversion
// ---------------------------------------------------------------------------
__global__ void cvt_f32_bf16(const float* __restrict__ in, bf16* __restrict__ out, int n) {
  int i = blockIdx.x * blockDim.x + threadIdx.x;
  if (i < n) out[i] = (bf16)in[i];
}

// ---------------------------------------------------------------------------
// Tiled bf16 WMMA GEMM:  C[M,N] = A[M,K] * B^T + bias   (B given as [N][K])
// mode 0: scatter into Q[H][T][64], K[H][T][64], Vt[H][64][T] (bf16)
// mode 1: write fp32 row-major out[M][N]
// Block: 256 threads (8 waves), tile 128(M) x 128(N) x 32(K).
// Double-buffered async-to-LDS staging (4 async B128 per wave per step).
// ---------------------------------------------------------------------------
#define BM 128
#define BN 128
#define BK 32
#define LDST 40   // LDS row stride in bf16 (80 bytes = 5*16: keeps 16B alignment)

__global__ __launch_bounds__(256)
void gemm_bf16_wmma(const bf16* __restrict__ A, const bf16* __restrict__ Bt,
                    const float* __restrict__ bias,
                    float* __restrict__ outF,
                    bf16* __restrict__ qb, bf16* __restrict__ kb, bf16* __restrict__ vT,
                    int M, int N, int K, int mode) {
  __shared__ bf16 As[2][BM * LDST];
  __shared__ bf16 Bs[2][BN * LDST];

  const int tid  = threadIdx.x;
  const int lane = tid & 31;
  const int wave = tid >> 5;
  const int bm0 = blockIdx.y * BM;
  const int bn0 = blockIdx.x * BN;
  const int wm = (wave & 3) * 32;   // 4 M sub-tiles of 32 rows
  const int wn = (wave >> 2) * 64;  // 2 N sub-tiles of 64 cols

  v8f acc[2][4];
  v8f vzero = {};
#pragma unroll
  for (int i = 0; i < 2; ++i)
#pragma unroll
    for (int j = 0; j < 4; ++j) acc[i][j] = vzero;

  // Per-thread staging slice: 16 bf16 (32B) of A and of B per step.
  const int lr = tid >> 1;          // 0..127: tile row loaded by this thread
  const int lc = (tid & 1) * 16;    // half-row: 16 bf16 = 32 bytes
  const bf16* gA = A  + (size_t)(bm0 + lr) * K + lc;
  const bf16* gB = Bt + (size_t)(bn0 + lr) * K + lc;
  const unsigned slot = (unsigned)(lr * LDST + lc);

  const int nsteps = K / BK;

  // Prologue: stage step 0 into buffer 0.
  {
    unsigned la = lds_off32(&As[0][slot]);
    unsigned lb = lds_off32(&Bs[0][slot]);
    async_b128(la,       gA);
    async_b128(la + 16u, gA + 8);
    async_b128(lb,       gB);
    async_b128(lb + 16u, gB + 8);
  }

  for (int i = 0; i < nsteps; ++i) {
    const int cur = i & 1;
    if (i + 1 < nsteps) {
      const int nk = (i + 1) * BK;
      const int nb = (i + 1) & 1;
      unsigned la = lds_off32(&As[nb][slot]);
      unsigned lb = lds_off32(&Bs[nb][slot]);
      async_b128(la,       gA + nk);
      async_b128(la + 16u, gA + nk + 8);
      async_b128(lb,       gB + nk);
      async_b128(lb + 16u, gB + nk + 8);
      // 4 newest in flight; oldest 4 (current step) must have landed.
      asm volatile("s_wait_asynccnt 0x4" ::: "memory");
    } else {
      asm volatile("s_wait_asynccnt 0x0" ::: "memory");
    }
    __syncthreads();   // current buffer visible to all waves

    v16bf af[2], bf[4];
#pragma unroll
    for (int ii = 0; ii < 2; ++ii) af[ii] = ldfrag(&As[cur][(wm + ii * 16) * LDST], LDST);
#pragma unroll
    for (int jj = 0; jj < 4; ++jj) bf[jj] = ldfrag(&Bs[cur][(wn + jj * 16) * LDST], LDST);

#pragma unroll
    for (int ii = 0; ii < 2; ++ii)
#pragma unroll
      for (int jj = 0; jj < 4; ++jj)
        acc[ii][jj] = wmma_bf16(af[ii], bf[jj], acc[ii][jj]);

    __syncthreads();   // all waves done reading before this buffer is re-staged
  }

  // Epilogue. C layout: element v -> row = v + 8*(lane>>4), col = lane&15.
  const int rsel = (lane >> 4) * 8;
  const int nsel = lane & 15;
#pragma unroll
  for (int i = 0; i < 2; ++i) {
#pragma unroll
    for (int j = 0; j < 4; ++j) {
      const int gn = bn0 + wn + j * 16 + nsel;
      const float bv = bias[gn];
#pragma unroll
      for (int v = 0; v < 8; ++v) {
        const int gm = bm0 + wm + i * 16 + rsel + v;
        const float val = acc[i][j][v] + bv;
        if (mode == 1) {
          outF[(size_t)gm * N + gn] = val;
        } else {
          if (gn < D_MODEL) {                 // Q
            int h = gn >> 6, d = gn & 63;
            qb[((size_t)h * M + gm) * HEAD_DIM + d] = (bf16)val;
          } else if (gn < 2 * D_MODEL) {      // K
            int g = gn - D_MODEL;
            int h = g >> 6, d = g & 63;
            kb[((size_t)h * M + gm) * HEAD_DIM + d] = (bf16)val;
          } else {                            // V, stored transposed [H][64][T]
            int g = gn - 2 * D_MODEL;
            int h = g >> 6, d = g & 63;
            vT[((size_t)(h << 6) + d) * M + gm] = (bf16)val;
          }
        }
      }
    }
  }
}

// ---------------------------------------------------------------------------
// RoPE applied in place to Q and K, layout [H][T][64].
// out[j]    = x[j+32]*cos + x[j]*sin       (j in [0,32))
// out[j+32] = x[j]*cos    - x[j+32]*sin ,  theta_j = 10000^(j/31), ang = t/theta
// ---------------------------------------------------------------------------
__global__ void rope_kernel(bf16* __restrict__ qb, bf16* __restrict__ kb, int T) {
  const int total = N_HEADS * T * 32;
  int idx = blockIdx.x * blockDim.x + threadIdx.x;
  if (idx >= 2 * total) return;
  bf16* buf = (idx < total) ? qb : kb;
  int id = (idx < total) ? idx : idx - total;
  int h = id / (T * 32);
  int rem = id - h * (T * 32);
  int t = rem >> 5;
  int j = rem & 31;
  bf16* p = buf + ((size_t)h * T + t) * HEAD_DIM;
  float xh = (float)p[j];        // first half
  float xl = (float)p[j + 32];   // second half
  float theta = __powf(10000.0f, (float)j * (1.0f / 31.0f));
  float ang = (float)t / theta;
  float sn = __sinf(ang), cs = __cosf(ang);
  p[j]      = (bf16)(xl * cs + xh * sn);
  p[j + 32] = (bf16)(xh * cs - xl * sn);
}

// ---------------------------------------------------------------------------
// Causal flash attention. Grid: (T/64, H). Block: 128 threads = 4 waves.
// Wave w handles Q rows [blockIdx.x*64 + w*16, +16) of head blockIdx.y.
// Key blocks of 32. All matmuls v_wmma_f32_16x16x32_bf16.
// ---------------------------------------------------------------------------
__global__ __launch_bounds__(128)
void attn_kernel(const bf16* __restrict__ qb, const bf16* __restrict__ kb,
                 const bf16* __restrict__ vT, bf16* __restrict__ yb, int T) {
  __shared__ bf16 plds[4][16 * LDST];   // per-wave P tile (16x32, stride 40)

  const int lane = threadIdx.x & 31;
  const int wave = threadIdx.x >> 5;
  const int h = blockIdx.y;
  const int q0 = blockIdx.x * 64 + wave * 16;
  const float scale = 0.125f;           // 1/sqrt(64)

  const bf16* qbase = qb + ((size_t)h * T + q0) * HEAD_DIM;
  const v16bf aq0 = ldfrag(qbase, HEAD_DIM);
  const v16bf aq1 = ldfrag(qbase + 32, HEAD_DIM);

  v8f acc[4];
  v8f vzero = {};
#pragma unroll
  for (int j = 0; j < 4; ++j) acc[j] = vzero;
  float mi[8], li[8];
#pragma unroll
  for (int v = 0; v < 8; ++v) { mi[v] = -1e30f; li[v] = 0.0f; }

  const int rbase = q0 + ((lane >> 4) * 8);  // global row of element v = rbase+v
  const int qmax = q0 + 15;

  for (int j0 = 0; j0 <= qmax; j0 += 32) {
    // Prefetch next key/value block (global_prefetch_b8).
    if (j0 + 32 <= qmax) {
      __builtin_prefetch(kb + ((size_t)h * T + j0 + 32 + lane) * HEAD_DIM, 0, 3);
      __builtin_prefetch(vT + ((size_t)(h << 6) + (lane & 15)) * T + j0 + 32, 0, 3);
    }

    // ---- S = Q K^T (16x32 tile, two 16x16 fragments, K-dim 64 = 2 x 32) ----
    v8f s[2];
#pragma unroll
    for (int jj = 0; jj < 2; ++jj) {
      const bf16* kbase = kb + ((size_t)h * T + j0 + jj * 16) * HEAD_DIM;
      v16bf bk0 = ldfrag(kbase, HEAD_DIM);
      v16bf bk1 = ldfrag(kbase + 32, HEAD_DIM);
      v8f z = vzero;
      z = wmma_bf16(aq0, bk0, z);
      z = wmma_bf16(aq1, bk1, z);
      s[jj] = z;
    }

    // ---- causal mask + scale, row max ----
    const int c0 = j0 + (lane & 15);
    float rowmax[8];
#pragma unroll
    for (int v = 0; v < 8; ++v) {
      const int r = rbase + v;
      float s0 = (c0 <= r)      ? s[0][v] * scale : -1e30f;
      float s1 = (c0 + 16 <= r) ? s[1][v] * scale : -1e30f;
      s[0][v] = s0; s[1][v] = s1;
      rowmax[v] = fmaxf(s0, s1);
    }
#pragma unroll
    for (int off = 8; off > 0; off >>= 1)
#pragma unroll
      for (int v = 0; v < 8; ++v)
        rowmax[v] = fmaxf(rowmax[v], __shfl_xor(rowmax[v], off, 32));

    // ---- online softmax update ----
    float corr[8], rowsum[8];
#pragma unroll
    for (int v = 0; v < 8; ++v) {
      float mnew = fmaxf(mi[v], rowmax[v]);
      corr[v] = __expf(mi[v] - mnew);
      mi[v] = mnew;
      float p0 = __expf(s[0][v] - mnew);
      float p1 = __expf(s[1][v] - mnew);
      s[0][v] = p0; s[1][v] = p1;
      rowsum[v] = p0 + p1;
    }
#pragma unroll
    for (int off = 8; off > 0; off >>= 1)
#pragma unroll
      for (int v = 0; v < 8; ++v)
        rowsum[v] += __shfl_xor(rowsum[v], off, 32);
#pragma unroll
    for (int v = 0; v < 8; ++v) li[v] = li[v] * corr[v] + rowsum[v];

    // ---- P: C layout -> A layout via per-wave LDS bounce ----
    bf16* pl = &plds[wave][0];
#pragma unroll
    for (int v = 0; v < 8; ++v) {
      const int row = v + ((lane >> 4) * 8);
      pl[row * LDST + (lane & 15)]      = (bf16)s[0][v];
      pl[row * LDST + 16 + (lane & 15)] = (bf16)s[1][v];
    }
    const v16bf ap = ldfrag(pl, LDST);

    // ---- O = O*corr + P V  (V^T rows are contiguous: vT[H][64][T]) ----
#pragma unroll
    for (int jd = 0; jd < 4; ++jd) {
      const bf16* vbase = vT + ((size_t)(h << 6) + jd * 16) * T + j0;
      v16bf bv = ldfrag(vbase, T);
#pragma unroll
      for (int v = 0; v < 8; ++v) acc[jd][v] *= corr[v];
      acc[jd] = wmma_bf16(ap, bv, acc[jd]);
    }
  }

  // ---- normalize, write y[T][H*64] bf16 ----
#pragma unroll
  for (int jd = 0; jd < 4; ++jd) {
#pragma unroll
    for (int v = 0; v < 8; ++v) {
      const int r = rbase + v;
      const float o = acc[jd][v] / li[v];
      yb[(size_t)r * D_MODEL + (h << 6) + jd * 16 + (lane & 15)] = (bf16)o;
    }
  }
}

// ---------------------------------------------------------------------------
// Host-side launch
// ---------------------------------------------------------------------------
extern "C" void kernel_launch(void* const* d_in, const int* in_sizes, int n_in,
                              void* d_out, int out_size, void* d_ws, size_t ws_size,
                              hipStream_t stream) {
  const float* x     = (const float*)d_in[0];
  const float* W_in  = (const float*)d_in[1];
  const float* b_in  = (const float*)d_in[2];
  const float* W_out = (const float*)d_in[3];
  const float* b_out = (const float*)d_in[4];
  float* out = (float*)d_out;

  const int T = T_LEN, D = D_MODEL, H = N_HEADS;

  char* ws = (char*)d_ws;
  size_t off = 0;
  auto alloc = [&](size_t bytes) -> void* {
    void* p = ws + off;
    off += (bytes + 255) & ~(size_t)255;
    return p;
  };
  bf16* xb  = (bf16*)alloc((size_t)T * D * 2);            // x bf16
  bf16* wb  = (bf16*)alloc((size_t)NQKV * D * 2);         // W_in bf16
  bf16* wob = (bf16*)alloc((size_t)D * D * 2);            // W_out bf16
  bf16* qb  = (bf16*)alloc((size_t)H * T * HEAD_DIM * 2); // Q [H][T][64]
  bf16* kb  = (bf16*)alloc((size_t)H * T * HEAD_DIM * 2); // K [H][T][64]
  bf16* vT  = (bf16*)alloc((size_t)H * HEAD_DIM * T * 2); // V^T [H][64][T]
  bf16* yb  = (bf16*)alloc((size_t)T * D * 2);            // attention out

  // 1) converts
  {
    int n = T * D;
    cvt_f32_bf16<<<(n + 255) / 256, 256, 0, stream>>>(x, xb, n);
    n = NQKV * D;
    cvt_f32_bf16<<<(n + 255) / 256, 256, 0, stream>>>(W_in, wb, n);
    n = D * D;
    cvt_f32_bf16<<<(n + 255) / 256, 256, 0, stream>>>(W_out, wob, n);
  }

  // 2) QKV projection GEMM (M=T, N=3072, K=1024), scatter epilogue
  {
    dim3 grid(NQKV / BN, T / BM);
    gemm_bf16_wmma<<<grid, 256, 0, stream>>>(xb, wb, b_in, nullptr,
                                             qb, kb, vT, T, NQKV, D, 0);
  }

  // 3) RoPE on Q and K
  {
    int n = 2 * H * T * 32;
    rope_kernel<<<(n + 255) / 256, 256, 0, stream>>>(qb, kb, T);
  }

  // 4) causal flash attention
  {
    dim3 grid(T / 64, H);
    attn_kernel<<<grid, 128, 0, stream>>>(qb, kb, vT, yb, T);
  }

  // 5) output GEMM (M=T, N=1024, K=1024), fp32 out + bias
  {
    dim3 grid(D / BN, T / BM);
    gemm_bf16_wmma<<<grid, 256, 0, stream>>>(yb, wob, b_out, out,
                                             nullptr, nullptr, nullptr, T, D, D, 1);
  }
}
